// GenericMPBaseline_15736760172599
// MI455X (gfx1250) — matrix operand order repaired and checked by reading, not compiled, not run on hardware
//
#include <hip/hip_runtime.h>
#include <hip/hip_bf16.h>
#include <cmath>

// Problem constants (match reference)
#define VOCAB   256
#define DIM     64
#define PATCH   4
#define BB      32
#define TT      8192
#define NPB     (TT / PATCH)      // 2048 nodes per batch row (power of two)
#define NODES   (BB * NPB)        // 65536 total nodes

typedef __attribute__((ext_vector_type(16))) _Float16 v16h;
typedef __attribute__((ext_vector_type(4)))  _Float16 v4h;
typedef __attribute__((ext_vector_type(8)))  float    v8f;

// ---------------------------------------------------------------------------
// WMMA helper (CDNA5 wave32, 16x16x32 f16 -> f32)
// ---------------------------------------------------------------------------
__device__ __forceinline__ v8f wmma16(v16h a, v16h b, v8f c) {
  // (neg_a, A, neg_b, B, c_mod, C, reuse_a, reuse_b)
  return __builtin_amdgcn_wmma_f32_16x16x32_f16(false, a, false, b, (short)0, c,
                                                false, false);
}

// A fragment (16x32 f16, MxK), source row-major in LDS: a[m*ld + k].
// Lanes 0-15: row M=lane, K = kb+{0..7, 16..23}; lanes 16-31: K = kb+{8..15, 24..31}.
// Two contiguous 16-byte runs per lane -> 2x ds_load_b128.
__device__ __forceinline__ v16h load_a_frag(const _Float16* a, int ld, int kb,
                                            int lane) {
  int m  = lane & 15;
  int kh = (lane >> 4) * 8;
  const _Float16* row = a + m * ld + kb + kh;
  v16h r;
#pragma unroll
  for (int j = 0; j < 8; ++j) {
    r[j]     = row[j];
    r[j + 8] = row[16 + j];
  }
  return r;
}

// Packed-B layout: for a KxN weight, tile t = kc*(N/16) + nc covers rows
// kb=32*kc, cols nb=16*nc. Lane l holds column n = nb + (l&15), K values
// kb + ((l>>4)*8) + {0..7, 16..23} stored contiguously (32 bytes, aligned).
// One aligned v16h LDS load -> 2x ds_load_b128.
__device__ __forceinline__ v16h load_b_packed(const _Float16* w, int tile,
                                              int lane) {
  return *(const v16h*)(w + (size_t)(tile * 32 + lane) * 16);
}

// Cooperative pack: coalesced global f32 reads, scattered f16 LDS writes.
template <int K, int N>
__device__ __forceinline__ void pack_weights(const float* __restrict__ src,
                                             _Float16* __restrict__ dst,
                                             int tid) {
  constexpr int NC = N / 16;
#pragma unroll 4
  for (int idx = tid; idx < K * N; idx += 256) {
    int k = idx / N, n = idx % N;             // N is power of two -> shifts
    int t    = (k >> 5) * NC + (n >> 4);
    int lane = (n & 15) + (((k >> 3) & 1) << 4);
    dst[(t * 32 + lane) * 16 + ((k & 7) | (((k >> 4) & 1) << 3))] = (_Float16)src[idx];
  }
}

// Branchless GELU. erf via Winitzki approximation:
//   erf(u) ~= sgn(u) * sqrt(1 - exp(-u^2 (4/pi + a u^2)/(1 + a u^2))), a=0.147
// abs err ~1e-4 — below the f16 quantization already used for activations.
// All ops map to straight-line VALU + v_rcp/v_exp/v_sqrt (no exec divergence).
__device__ __forceinline__ float gelu_fast(float x) {
  float u   = x * 0.70710678118654752440f;
  float u2  = u * u;
  float num = u2 * __builtin_fmaf(0.147f, u2, 1.2732395447351628f);
  float den = __builtin_fmaf(0.147f, u2, 1.0f);
  float r   = -num * __builtin_amdgcn_rcpf(den);
  float e   = __expf(r);
  float t   = __builtin_amdgcn_sqrtf(__builtin_fmaxf(1.0f - e, 0.0f));
  float er  = __builtin_copysignf(t, x);
  return 0.5f * x * (1.0f + er);
}

// ---------------------------------------------------------------------------
// Kernel 1: embedding gather + patch mean  -> h [NODES, 64] f32
// ---------------------------------------------------------------------------
__global__ __launch_bounds__(256) void gmp_embed_mean(
    const int* __restrict__ x, const float* __restrict__ emb,
    float* __restrict__ h) {
  int t    = blockIdx.x * 256 + threadIdx.x;
  int node = t >> 4;
  int q    = (t & 15) * 4;  // dim offset, float4 aligned
  const int* xr = x + (size_t)node * PATCH;
  int i0 = xr[0], i1 = xr[1], i2 = xr[2], i3 = xr[3];
  const float4 e0 = *(const float4*)(emb + (size_t)i0 * DIM + q);
  const float4 e1 = *(const float4*)(emb + (size_t)i1 * DIM + q);
  const float4 e2 = *(const float4*)(emb + (size_t)i2 * DIM + q);
  const float4 e3 = *(const float4*)(emb + (size_t)i3 * DIM + q);
  float4 s;
  s.x = (e0.x + e1.x + e2.x + e3.x) * 0.25f;
  s.y = (e0.y + e1.y + e2.y + e3.y) * 0.25f;
  s.z = (e0.z + e1.z + e2.z + e3.z) * 0.25f;
  s.w = (e0.w + e1.w + e2.w + e3.w) * 0.25f;
  *(float4*)(h + (size_t)node * DIM + q) = s;
}

// ---------------------------------------------------------------------------
// Kernel 2: one message-passing round, fully fused.
// 256 threads = 8 waves; each wave owns a 16-node tile (block = 128 nodes).
// Static LDS ~ 58 KB.
// ---------------------------------------------------------------------------
__global__ __launch_bounds__(256) void gmp_round(
    const float* __restrict__ h_in, float* __restrict__ h_out,
    const float* __restrict__ wm1, const float* __restrict__ bm1,
    const float* __restrict__ wm2, const float* __restrict__ bm2,
    const float* __restrict__ wu1, const float* __restrict__ bu1,
    const float* __restrict__ wu2, const float* __restrict__ bu2,
    const float* __restrict__ lng, const float* __restrict__ lnb) {
  __shared__ _Float16 s_w1[128 * 64];      // 16 KB packed (wm1, then wu1)
  __shared__ _Float16 s_w2[64 * 64];       // 8 KB packed  (wm2, then wu2)
  __shared__ float    s_bias[4 * 64];      // bm1|bm2|bu1|bu2
  __shared__ float    s_ln[2 * 64];        // gamma|beta
  __shared__ _Float16 s_z[8][16 * 128];    // per-wave [h | left/msgs] f16, 32 KB

  const int tid = threadIdx.x;

  // Phase A weights (msg net), pre-packed into B-fragment layout
  pack_weights<128, 64>(wm1, s_w1, tid);
  pack_weights<64, 64>(wm2, s_w2, tid);
  if (tid < 64) {
    s_bias[tid]       = bm1[tid];
    s_bias[64 + tid]  = bm2[tid];
    s_bias[128 + tid] = bu1[tid];
    s_bias[192 + tid] = bu2[tid];
    s_ln[tid]         = lng[tid];
    s_ln[64 + tid]    = lnb[tid];
  }
  // warm the phase-B weights into cache while we compute phase A
  __builtin_prefetch(wu1 + tid * 8, 0, 3);
  __builtin_prefetch(wu2 + tid * 8, 0, 3);

  const int wave = tid >> 5, lane = tid & 31;
  const int n15 = lane & 15, mb = (lane >> 4) * 8;
  const int tile = blockIdx.x * 8 + wave;
  const int g0   = tile * 16;                 // first global node of this tile
  _Float16* z = s_z[wave];

  // load h tile (cols 0..63) and left-neighbor tile (cols 64..127) as f16
  {
    int m = n15, half = lane >> 4;
    int g = g0 + m;
    bool has_left = (g & (NPB - 1)) != 0;   // node 0 of a batch row has no left
    const float* hr = h_in + (size_t)g * DIM + half * 32;
    const float* lr =
        h_in + (size_t)(has_left ? g - 1 : g) * DIM + half * 32;  // safe ptr
    float lscale = has_left ? 1.0f : 0.0f;
#pragma unroll
    for (int j = 0; j < 32; j += 4) {
      float4 hv = *(const float4*)(hr + j);
      *(v4h*)&z[m * 128 + half * 32 + j] =
          (v4h){(_Float16)hv.x, (_Float16)hv.y, (_Float16)hv.z, (_Float16)hv.w};
      float4 lv = *(const float4*)(lr + j);
      *(v4h*)&z[m * 128 + 64 + half * 32 + j] =
          (v4h){(_Float16)(lv.x * lscale), (_Float16)(lv.y * lscale),
                (_Float16)(lv.z * lscale), (_Float16)(lv.w * lscale)};
    }
  }
  __syncthreads();

  // ================= msg = MLP([h, left]) =================
  {
    v16h a0 = load_a_frag(z, 128, 0, lane);
    v16h a1 = load_a_frag(z, 128, 32, lane);
    v16h a2 = load_a_frag(z, 128, 64, lane);
    v16h a3 = load_a_frag(z, 128, 96, lane);
#pragma unroll
    for (int c = 0; c < 4; ++c) {
      float bv = s_bias[c * 16 + n15];
      v8f d;
#pragma unroll
      for (int r = 0; r < 8; ++r) d[r] = bv;
      d = wmma16(a0, load_b_packed(s_w1, 0 * 4 + c, lane), d);
      d = wmma16(a1, load_b_packed(s_w1, 1 * 4 + c, lane), d);
      d = wmma16(a2, load_b_packed(s_w1, 2 * 4 + c, lane), d);
      d = wmma16(a3, load_b_packed(s_w1, 3 * 4 + c, lane), d);
      // GELU, stash hidden into z cols 64..127 (left part is dead: in regs)
#pragma unroll
      for (int r = 0; r < 8; ++r)
        z[(mb + r) * 128 + 64 + c * 16 + n15] = (_Float16)gelu_fast(d[r]);
    }
  }
  {
    // layer 2: read hidden (regs), overwrite same region with msgs
    v16h a0 = load_a_frag(z + 64, 128, 0, lane);
    v16h a1 = load_a_frag(z + 64, 128, 32, lane);
#pragma unroll
    for (int c = 0; c < 4; ++c) {
      float bv = s_bias[64 + c * 16 + n15];
      v8f d;
#pragma unroll
      for (int r = 0; r < 8; ++r) d[r] = bv;
      d = wmma16(a0, load_b_packed(s_w2, 0 * 4 + c, lane), d);
      d = wmma16(a1, load_b_packed(s_w2, 1 * 4 + c, lane), d);
#pragma unroll
      for (int r = 0; r < 8; ++r)
        z[(mb + r) * 128 + 64 + c * 16 + n15] = (_Float16)d[r];
    }
  }
  __syncthreads();

  // Phase B weights: update net (reuses the same LDS buffers)
  pack_weights<128, 64>(wu1, s_w1, tid);
  pack_weights<64, 64>(wu2, s_w2, tid);
  __syncthreads();

  // ================= upd = MLP([h, msgs]) =================
  v8f sv[4];  // h + upd in D-fragment layout, one per 16-col chunk
  {
    v16h a0 = load_a_frag(z, 128, 0, lane);
    v16h a1 = load_a_frag(z, 128, 32, lane);
    v16h a2 = load_a_frag(z, 128, 64, lane);
    v16h a3 = load_a_frag(z, 128, 96, lane);
#pragma unroll
    for (int c = 0; c < 4; ++c) {
      float bv = s_bias[128 + c * 16 + n15];
      v8f d;
#pragma unroll
      for (int r = 0; r < 8; ++r) d[r] = bv;
      d = wmma16(a0, load_b_packed(s_w1, 0 * 4 + c, lane), d);
      d = wmma16(a1, load_b_packed(s_w1, 1 * 4 + c, lane), d);
      d = wmma16(a2, load_b_packed(s_w1, 2 * 4 + c, lane), d);
      d = wmma16(a3, load_b_packed(s_w1, 3 * 4 + c, lane), d);
#pragma unroll
      for (int r = 0; r < 8; ++r)
        z[(mb + r) * 128 + 64 + c * 16 + n15] = (_Float16)gelu_fast(d[r]);
    }
  }
  {
    v16h a0 = load_a_frag(z + 64, 128, 0, lane);
    v16h a1 = load_a_frag(z + 64, 128, 32, lane);
#pragma unroll
    for (int c = 0; c < 4; ++c) {
      float bv = s_bias[192 + c * 16 + n15];
      v8f d;
#pragma unroll
      for (int r = 0; r < 8; ++r) d[r] = bv;
      d = wmma16(a0, load_b_packed(s_w2, 0 * 4 + c, lane), d);
      d = wmma16(a1, load_b_packed(s_w2, 1 * 4 + c, lane), d);
      // residual: reload h (f32) at D-layout; 64B coalesced per half-wave
#pragma unroll
      for (int r = 0; r < 8; ++r)
        d[r] += h_in[(size_t)(g0 + mb + r) * DIM + c * 16 + n15];
      sv[c] = d;
    }
  }

  // ================= LayerNorm over 64 features, in registers =============
  // Row m = mb + r lives in a 16-lane half; xor masks 1,2,4,8 stay in-half.
  float acc_s[8], acc_q[8];
#pragma unroll
  for (int r = 0; r < 8; ++r) {
    acc_s[r] = sv[0][r] + sv[1][r] + sv[2][r] + sv[3][r];
    acc_q[r] = sv[0][r] * sv[0][r] + sv[1][r] * sv[1][r] +
               sv[2][r] * sv[2][r] + sv[3][r] * sv[3][r];
  }
#pragma unroll
  for (int mask = 1; mask <= 8; mask <<= 1) {
#pragma unroll
    for (int r = 0; r < 8; ++r) {
      acc_s[r] += __shfl_xor(acc_s[r], mask, 32);
      acc_q[r] += __shfl_xor(acc_q[r], mask, 32);
    }
  }
#pragma unroll
  for (int r = 0; r < 8; ++r) {
    float mu  = acc_s[r] * (1.0f / 64.0f);
    float var = acc_q[r] * (1.0f / 64.0f) - mu * mu;
    float inv = rsqrtf(var + 1e-5f);
    float* orow = h_out + (size_t)(g0 + mb + r) * DIM;
#pragma unroll
    for (int c = 0; c < 4; ++c) {
      int n = c * 16 + n15;
      orow[n] = (sv[c][r] - mu) * inv * s_ln[n] + s_ln[64 + n];
    }
  }
}

// ---------------------------------------------------------------------------
// Kernel 3: readout  out[g*4 + p, :] = h[g] @ w_r + b_r  (p = 0..3)
// Output is write-once (268 MB): stream it with non-temporal stores so the
// h ping-pong (32 MB) stays resident in the 192 MB L2.
// ---------------------------------------------------------------------------
__global__ __launch_bounds__(256) void gmp_readout(
    const float* __restrict__ h, const float* __restrict__ wr,
    const float* __restrict__ br, float* __restrict__ out) {
  __shared__ _Float16 s_w[64 * VOCAB];   // 32 KB packed, tiles: kc*16 + nc
  __shared__ float    s_br[VOCAB];
  __shared__ _Float16 s_z[8][16 * 64];   // 16 KB

  const int tid = threadIdx.x;
  pack_weights<64, VOCAB>(wr, s_w, tid);
  if (tid < VOCAB) s_br[tid] = br[tid];

  const int wave = tid >> 5, lane = tid & 31;
  const int n15 = lane & 15, mb = (lane >> 4) * 8;
  const int tile = blockIdx.x * 8 + wave;
  const int g0   = tile * 16;
  _Float16* z = s_z[wave];
  {
    int m = n15, half = lane >> 4;
    const float* hr = h + (size_t)(g0 + m) * DIM + half * 32;
#pragma unroll
    for (int j = 0; j < 32; j += 4) {
      float4 hv = *(const float4*)(hr + j);
      *(v4h*)&z[m * 64 + half * 32 + j] =
          (v4h){(_Float16)hv.x, (_Float16)hv.y, (_Float16)hv.z, (_Float16)hv.w};
    }
  }
  __syncthreads();

  v16h a0 = load_a_frag(z, 64, 0, lane);
  v16h a1 = load_a_frag(z, 64, 32, lane);
#pragma unroll
  for (int c = 0; c < 16; ++c) {
    float bv = s_br[c * 16 + n15];
    v8f d;
#pragma unroll
    for (int r = 0; r < 8; ++r) d[r] = bv;
    d = wmma16(a0, load_b_packed(s_w, 0 * 16 + c, lane), d);
    d = wmma16(a1, load_b_packed(s_w, 1 * 16 + c, lane), d);
#pragma unroll
    for (int r = 0; r < 8; ++r) {
      size_t node = (size_t)(g0 + mb + r);
      float* o = out + node * 4 * VOCAB + c * 16 + n15;
      float v = d[r];
      __builtin_nontemporal_store(v, o);
      __builtin_nontemporal_store(v, o + VOCAB);
      __builtin_nontemporal_store(v, o + 2 * VOCAB);
      __builtin_nontemporal_store(v, o + 3 * VOCAB);
    }
  }
}

// ---------------------------------------------------------------------------
extern "C" void kernel_launch(void* const* d_in, const int* in_sizes, int n_in,
                              void* d_out, int out_size, void* d_ws,
                              size_t ws_size, hipStream_t stream) {
  (void)in_sizes; (void)n_in; (void)out_size; (void)ws_size;
  const int*   x   = (const int*)d_in[0];
  const float* emb = (const float*)d_in[1];
  const float* wm1 = (const float*)d_in[2];
  const float* bm1 = (const float*)d_in[3];
  const float* wm2 = (const float*)d_in[4];
  const float* bm2 = (const float*)d_in[5];
  const float* wu1 = (const float*)d_in[6];
  const float* bu1 = (const float*)d_in[7];
  const float* wu2 = (const float*)d_in[8];
  const float* bu2 = (const float*)d_in[9];
  const float* lng = (const float*)d_in[10];
  const float* lnb = (const float*)d_in[11];
  const float* wr  = (const float*)d_in[12];
  const float* br  = (const float*)d_in[13];
  float* out = (float*)d_out;

  float* h0 = (float*)d_ws;                       // 16 MB
  float* h1 = h0 + (size_t)NODES * DIM;           // 16 MB

  gmp_embed_mean<<<NODES * 16 / 256, 256, 0, stream>>>(x, emb, h0);

  const int rblocks = NODES / (16 * 8);           // 512
  gmp_round<<<rblocks, 256, 0, stream>>>(h0, h1, wm1, bm1, wm2, bm2,
                                         wu1, bu1, wu2, bu2, lng, lnb);
  gmp_round<<<rblocks, 256, 0, stream>>>(h1, h0, wm1, bm1, wm2, bm2,
                                         wu1, bu1, wu2, bu2, lng, lnb);
  gmp_round<<<rblocks, 256, 0, stream>>>(h0, h1, wm1, bm1, wm2, bm2,
                                         wu1, bu1, wu2, bu2, lng, lnb);

  gmp_readout<<<rblocks, 256, 0, stream>>>(h1, wr, br, out);
}
